// SyntacticGCN_23484881174529
// MI455X (gfx1250) — matrix operand order repaired and checked by reading, not compiled, not run on hardware
//
#include <hip/hip_runtime.h>
#include <hip/hip_bf16.h>

// ================= CDNA5 (gfx1250) wave32 types ============================
typedef __attribute__((ext_vector_type(16))) __bf16 v16bf;
typedef __attribute__((ext_vector_type(8)))  __bf16 v8bf;
typedef __attribute__((ext_vector_type(8)))  float  v8f;
typedef __attribute__((ext_vector_type(4)))  unsigned u32x4;
typedef __attribute__((ext_vector_type(8)))  unsigned u32x8;

#define BB    4
#define NN    256
#define DIN0  400
#define DMEM  200
// padded K (mult of 32) per layer input width, padded N tile counts
#define KP0   416   // pad(400)
#define KP1   224   // pad(200)
#define NNT0  25    // ceil(400/16)
#define NNT1  13    // ceil(200/16) -> 208
#define NKT0  13    // KP0/32
#define NKT1  7     // KP1/32
#define NNTW  13    // weight output tiles (200 -> 208)

// ================= Tensor Data Mover (TDM) =================================
// D# per CDNA5 ISA.  2-D tile: tile_d0 x tile_d1
// elements (data_size units) from a tensor with row stride stride0, copied to
// contiguous LDS at lds_off.  Issued via inline asm (portable across both
// toolchains); completion via TENSORcnt.
__device__ __forceinline__ void tdm_load_2d(unsigned lds_off, const void* gptr,
                                            unsigned tile_d0, unsigned tile_d1,
                                            unsigned tensor_d0, unsigned tensor_d1,
                                            unsigned long long stride0,
                                            unsigned data_size_code) {
  unsigned long long ga = (unsigned long long)gptr;
  u32x4 g0;
  g0[0] = 1u;                                            // count=1 (valid user D#)
  g0[1] = lds_off;                                       // lds_addr
  g0[2] = (unsigned)(ga & 0xffffffffull);                // global_addr[31:0]
  g0[3] = (unsigned)((ga >> 32) & 0x01ffffffull)         // global_addr[56:32]
        | (2u << 30);                                    // type=2 ("image")
  u32x8 g1;
  g1[0] = (data_size_code & 3u) << 16;                   // data_size; mask=0
  g1[1] = (tensor_d0 & 0xffffu) << 16;                   // tensor_dim0[15:0]
  g1[2] = (tensor_d0 >> 16) | ((tensor_d1 & 0xffffu) << 16);
  g1[3] = (tensor_d1 >> 16) | ((tile_d0 & 0xffffu) << 16);
  g1[4] = (tile_d1 & 0xffffu);                           // tile_dim1; tile_dim2=0
  g1[5] = (unsigned)(stride0 & 0xffffffffull);           // dim0_stride[31:0]
  g1[6] = (unsigned)((stride0 >> 32) & 0xffffull);       // dim0_stride[47:32]
  g1[7] = 0u;
  u32x4 gz = {0u, 0u, 0u, 0u};                           // groups 2/3: dims unused
  asm volatile("tensor_load_to_lds %0, %1, %2, %3"
               :: "s"(g0), "s"(g1), "s"(gz), "s"(gz) : "memory");
}
__device__ __forceinline__ void tdm_wait() {
  __builtin_amdgcn_s_wait_tensorcnt(0);
}
__device__ __forceinline__ unsigned lds_off_of(const void* p) {
  return (unsigned)(unsigned long long)p;   // generic addr[31:0] == LDS offset
}

// ================= WMMA fragments (CDNA5 ISA 7.12.2) ==================
// A 16x32 bf16, this code's contiguous k order (not the ISA table order; B packed to match): lane<16 -> M=lane, K=h*16..+15, h=lane>>4.
// A fragment from an LDS row-block (16 rows x strideE elems): 2x ds_load_b128.
__device__ __forceinline__ v16bf frag_a_lds(const __bf16* lds, int strideE,
                                            int sub, int halfI, int k0) {
  const __bf16* p = lds + sub * strideE + k0 + halfI * 16;
  v8bf a0 = *(const v8bf*)p;
  v8bf a1 = *(const v8bf*)(p + 8);
  return __builtin_shufflevector(a0, a1, 0,1,2,3,4,5,6,7,8,9,10,11,12,13,14,15);
}
// B 32x16 bf16 fragment pre-packed as [lane][e] (32B/lane): 2x global b128.
__device__ __forceinline__ v16bf frag_b_pk(const __bf16* tile, int lane) {
  const __bf16* p = tile + lane * 16;
  v8bf b0 = *(const v8bf*)p;
  v8bf b1 = *(const v8bf*)(p + 8);
  return __builtin_shufflevector(b0, b1, 0,1,2,3,4,5,6,7,8,9,10,11,12,13,14,15);
}
__device__ __forceinline__ v8f wmma_bf16(v16bf a, v16bf b, v8f c) {
  return __builtin_amdgcn_wmma_f32_16x16x32_bf16(false, a, false, b, (short)0, c,
                                                 false, false);
}

// ================= prep kernels ============================================
__global__ void cvt_bf16_kernel(const float* __restrict__ s, __bf16* __restrict__ d, int n) {
  int i = blockIdx.x * 256 + threadIdx.x;
  if (i < n) d[i] = (__bf16)s[i];
}
// rows x K fp32 -> rows x KP bf16, zero-padded columns
__global__ void cvt_pad_kernel(const float* __restrict__ s, __bf16* __restrict__ d,
                               int K, int KP, int n) {
  int i = blockIdx.x * 256 + threadIdx.x;
  if (i >= n) return;
  int row = i / KP, col = i - row * KP;
  d[i] = (col < K) ? (__bf16)s[(long)row * K + col] : (__bf16)0.f;
}
__global__ void fill0_bf16_kernel(__bf16* __restrict__ d, int n) {
  int i = blockIdx.x * 256 + threadIdx.x;
  if (i < n) d[i] = (__bf16)0.f;
}
// pack x (per-batch 256 rows x KP cols, bf16) into B-fragment layout
// dst[(((b*8+kt)*NNT+nt)*32+lane)*16+e] = x[b*256 + kt*32+(lane>>4)*16+e][nt*16+(lane&15)]
__global__ void pack_x_kernel(const __bf16* __restrict__ xbf, __bf16* __restrict__ xp,
                              int KP, int NNT) {
  int i = blockIdx.x * 256 + threadIdx.x;
  int total = BB * 8 * NNT * 512;
  if (i >= total) return;
  int e = i & 15, lane = (i >> 4) & 31, t = i >> 9;
  int nt = t % NNT; t /= NNT;
  int kt = t & 7;   int b = t >> 3;
  int k = kt * 32 + (lane >> 4) * 16 + e;
  int n = nt * 16 + (lane & 15);
  xp[i] = xbf[((long)(b * NN + k)) * KP + n];
}
// pack W (200 x din, row-major fp32) as B(k,n)=W[n][k] into fragment layout
__global__ void pack_w_kernel(const float* __restrict__ W, __bf16* __restrict__ wp,
                              int din, int NKT) {
  int i = blockIdx.x * 256 + threadIdx.x;
  int total = NKT * NNTW * 512;
  if (i >= total) return;
  int e = i & 15, lane = (i >> 4) & 31, t = i >> 9;
  int nt = t % NNTW, kt = t / NNTW;
  int k = kt * 32 + (lane >> 4) * 16 + e;
  int n = nt * 16 + (lane & 15);
  wp[i] = (k < din && n < DMEM) ? (__bf16)W[(long)n * din + k] : (__bf16)0.f;
}

// ================= denom ===================================================
__global__ void gcn_denom_kernel(const float* __restrict__ adj_in,
                                 const float* __restrict__ adj_out,
                                 float* __restrict__ denom) {
  int i = blockIdx.x * 256 + threadIdx.x;
  if (i >= BB * NN) return;
  const float* r1 = adj_in  + (long)i * NN;
  const float* r2 = adj_out + (long)i * NN;
  float s = 1.0f;
  for (int j = 0; j < NN; ++j) s += r1[j] + r2[j];
  denom[i] = s;
}

// ================= adjacency GEMM: Ax and Bx together ======================
// grid (NNT, 16, BB), block 32.  A tiles staged to LDS by TDM.
template <int KP, int NNT>
__global__ void __launch_bounds__(32)
gcn_adj_gemm_kernel(const __bf16* __restrict__ adjin_bf,
                    const __bf16* __restrict__ adjout_bf,
                    const __bf16* __restrict__ xpack,
                    __bf16* __restrict__ Axbf, __bf16* __restrict__ Bxbf) {
  __shared__ __bf16 ldsA[2][16 * NN];
  const int b = blockIdx.z, tm = blockIdx.y, tn = blockIdx.x;
  const int lane = threadIdx.x & 31, halfI = lane >> 4, sub = lane & 15;

  tdm_load_2d(lds_off_of(&ldsA[0][0]), adjin_bf  + ((long)b * NN + tm * 16) * NN,
              NN, 16, NN, NN, NN, /*2B*/1u);
  tdm_load_2d(lds_off_of(&ldsA[1][0]), adjout_bf + ((long)b * NN + tm * 16) * NN,
              NN, 16, NN, NN, NN, 1u);
  tdm_wait();

  const __bf16* xp = xpack + (((long)b * 8) * NNT + tn) * 512;
  v8f accA = {}, accB = {};
#pragma unroll
  for (int kt = 0; kt < 8; ++kt) {
    v16bf fB = frag_b_pk(xp + (long)kt * NNT * 512, lane);
    v16bf aA = frag_a_lds(&ldsA[0][0], NN, sub, halfI, kt * 32);
    v16bf aO = frag_a_lds(&ldsA[1][0], NN, sub, halfI, kt * 32);
    accA = wmma_bf16(aA, fB, accA);
    accB = wmma_bf16(aO, fB, accB);
  }
  const long colo = tn * 16 + sub;
  __bf16* yA = Axbf + ((long)b * NN + tm * 16) * KP + colo;
  __bf16* yB = Bxbf + ((long)b * NN + tm * 16) * KP + colo;
#pragma unroll
  for (int r = 0; r < 8; ++r) {
    long m = halfI * 8 + r;
    yA[m * KP] = (__bf16)accA[r];
    yB[m * KP] = (__bf16)accB[r];
  }
}

// ================= fused 6-way linear (res + gres) =========================
// grid (13, 64), block 32.  Ax/Bx/X row-blocks staged to LDS by TDM.
template <int KP, int NKT>
__global__ void __launch_bounds__(32)
gcn_lin3_kernel(const __bf16* __restrict__ Axbf, const __bf16* __restrict__ Bxbf,
                const __bf16* __restrict__ Xbf,
                const __bf16* __restrict__ wp0, const __bf16* __restrict__ wp1,
                const __bf16* __restrict__ wp2,
                const __bf16* __restrict__ gp0, const __bf16* __restrict__ gp1,
                const __bf16* __restrict__ gp2,
                const float* __restrict__ b0, const float* __restrict__ b1,
                const float* __restrict__ b2,
                const float* __restrict__ gb0, const float* __restrict__ gb1,
                const float* __restrict__ gb2,
                const float* __restrict__ denom,
                float* __restrict__ resPre, float* __restrict__ gresPre) {
  __shared__ __bf16 ldsA[3][16 * KP];
  const int tm = blockIdx.y, tn = blockIdx.x;
  const int lane = threadIdx.x & 31, halfI = lane >> 4, sub = lane & 15;

  tdm_load_2d(lds_off_of(&ldsA[0][0]), Axbf + (long)tm * 16 * KP, KP, 16, KP,
              BB * NN, KP, 1u);
  tdm_load_2d(lds_off_of(&ldsA[1][0]), Bxbf + (long)tm * 16 * KP, KP, 16, KP,
              BB * NN, KP, 1u);
  tdm_load_2d(lds_off_of(&ldsA[2][0]), Xbf  + (long)tm * 16 * KP, KP, 16, KP,
              BB * NN, KP, 1u);
  tdm_wait();

  v8f aR = {}, aG = {};
#pragma unroll
  for (int kt = 0; kt < NKT; ++kt) {
    const long off = ((long)kt * NNTW + tn) * 512;
    v16bf fA = frag_a_lds(&ldsA[0][0], KP, sub, halfI, kt * 32);
    v16bf fB = frag_a_lds(&ldsA[1][0], KP, sub, halfI, kt * 32);
    v16bf fX = frag_a_lds(&ldsA[2][0], KP, sub, halfI, kt * 32);
    aR = wmma_bf16(fA, frag_b_pk(wp0 + off, lane), aR);
    aR = wmma_bf16(fB, frag_b_pk(wp1 + off, lane), aR);
    aR = wmma_bf16(fX, frag_b_pk(wp2 + off, lane), aR);
    aG = wmma_bf16(fA, frag_b_pk(gp0 + off, lane), aG);
    aG = wmma_bf16(fB, frag_b_pk(gp1 + off, lane), aG);
    aG = wmma_bf16(fX, frag_b_pk(gp2 + off, lane), aG);
  }
  const int n = tn * 16 + sub;
  if (n < DMEM) {
    const float bvR = b0[n] + b1[n] + b2[n];
    const float bvG = gb0[n] + gb1[n] + gb2[n];
#pragma unroll
    for (int r = 0; r < 8; ++r) {
      const int m = tm * 16 + halfI * 8 + r;
      const float dn = denom[m];
      resPre [(long)m * DMEM + n] = (aR[r] + bvR) / dn;
      gresPre[(long)m * DMEM + n] = (aG[r] + bvG) / dn;
    }
  }
}

// ================= sparse label-embedding bias =============================
__global__ void gcn_bias_emb_kernel(const float* __restrict__ adj_in,
                                    const float* __restrict__ adj_out,
                                    const int* __restrict__ dep,
                                    const float* __restrict__ emb,
                                    const float* __restrict__ gemb,
                                    float* __restrict__ biasR, float* __restrict__ biasG) {
  const int t = blockIdx.x, b = blockIdx.y, d = threadIdx.x;
  const long base = (long)b * NN * NN;
  float accR = 0.f, accG = 0.f;
  for (int s = 0; s < NN; ++s) {
    const float v1 = adj_in [base + (long)s * NN + t];
    const float v2 = adj_out[base + (long)t * NN + s];
    if (v1 != 0.f) {
      const int l = dep[base + (long)s * NN + t];
      if (d < DMEM) { accR += v1 * emb[(long)l * DMEM + d]; accG += v1 * gemb[(long)l * DMEM + d]; }
    }
    if (v2 != 0.f) {
      const int l = dep[base + (long)t * NN + s];
      if (d < DMEM) { accR += v2 * emb[(long)l * DMEM + d]; accG += v2 * gemb[(long)l * DMEM + d]; }
    }
  }
  if (d < DMEM) {
    const long o = ((long)b * NN + t) * DMEM + d;
    biasR[o] = accR;
    biasG[o] = accG;
  }
}

// ================= gate: x = relu(sigmoid(gres)*res) -> padded bf16 ========
__global__ void gcn_gate_kernel(const float* __restrict__ resPre,
                                const float* __restrict__ gresPre,
                                const float* __restrict__ biasR,
                                const float* __restrict__ biasG,
                                __bf16* __restrict__ xbf) {   // 1024 x KP1
  int i = blockIdx.x * 256 + threadIdx.x;
  if (i >= BB * NN * KP1) return;
  int row = i / KP1, col = i - row * KP1;
  __bf16 out = (__bf16)0.f;
  if (col < DMEM) {
    long j = (long)row * DMEM + col;
    float r = resPre[j] + biasR[j];
    float g = gresPre[j] + biasG[j];
    float sg = 1.f / (1.f + __expf(-g));
    float v = sg * r;
    out = (__bf16)(v > 0.f ? v : 0.f);
  }
  xbf[i] = out;
}

// ================= final MLP: relu(x @ Wmlp^T + b) -> d_out ================
__global__ void __launch_bounds__(32)
gcn_mlp_kernel(const __bf16* __restrict__ Xbf, const __bf16* __restrict__ wp,
               const float* __restrict__ bias, float* __restrict__ Out) {
  __shared__ __bf16 ldsA[16 * KP1];
  const int tm = blockIdx.y, tn = blockIdx.x;
  const int lane = threadIdx.x & 31, halfI = lane >> 4, sub = lane & 15;
  tdm_load_2d(lds_off_of(&ldsA[0]), Xbf + (long)tm * 16 * KP1, KP1, 16, KP1,
              BB * NN, KP1, 1u);
  tdm_wait();
  v8f acc = {};
#pragma unroll
  for (int kt = 0; kt < NKT1; ++kt) {
    v16bf fA = frag_a_lds(&ldsA[0], KP1, sub, halfI, kt * 32);
    acc = wmma_bf16(fA, frag_b_pk(wp + ((long)kt * NNTW + tn) * 512, lane), acc);
  }
  const int n = tn * 16 + sub;
  if (n < DMEM) {
    const float bv = bias[n];
#pragma unroll
    for (int r = 0; r < 8; ++r) {
      const int m = tm * 16 + halfI * 8 + r;
      const float v = acc[r] + bv;
      Out[(long)m * DMEM + n] = v > 0.f ? v : 0.f;
    }
  }
}

// ================= host orchestration ======================================
static inline int ceil_div(int a, int b) { return (a + b - 1) / b; }

extern "C" void kernel_launch(void* const* d_in, const int* in_sizes, int n_in,
                              void* d_out, int out_size, void* d_ws, size_t ws_size,
                              hipStream_t stream) {
  const float* gcn_inputs = (const float*)d_in[0];
  const float* adj_in     = (const float*)d_in[2];
  const float* adj_out    = (const float*)d_in[3];
  const int*   dep        = (const int*)d_in[4];
  const float* w_mlp      = (const float*)d_in[5 + 2 * 14 + 0];
  const float* b_mlp      = (const float*)d_in[5 + 2 * 14 + 1];

  // ---- workspace carve-out (bytes) ----
  char* base = (char*)d_ws;
  size_t off = 0;
  auto alloc = [&](size_t bytes) { void* p = base + off; off += (bytes + 255) & ~(size_t)255; return p; };
  __bf16* adjin_bf  = (__bf16*)alloc((size_t)BB * NN * NN * 2);
  __bf16* adjout_bf = (__bf16*)alloc((size_t)BB * NN * NN * 2);
  __bf16* xbf       = (__bf16*)alloc((size_t)BB * NN * KP0 * 2);
  __bf16* xpack     = (__bf16*)alloc((size_t)BB * 8 * NNT0 * 512 * 2);
  __bf16* Axbf      = (__bf16*)alloc((size_t)BB * NN * KP0 * 2);
  __bf16* Bxbf      = (__bf16*)alloc((size_t)BB * NN * KP0 * 2);
  __bf16* wp[6];
  for (int i = 0; i < 6; ++i) wp[i] = (__bf16*)alloc((size_t)NKT0 * NNTW * 512 * 2);
  __bf16* wmlp_p    = (__bf16*)alloc((size_t)NKT1 * NNTW * 512 * 2);
  float* resPre  = (float*)alloc((size_t)BB * NN * DMEM * 4);
  float* gresPre = (float*)alloc((size_t)BB * NN * DMEM * 4);
  float* biasR   = (float*)alloc((size_t)BB * NN * DMEM * 4);
  float* biasG   = (float*)alloc((size_t)BB * NN * DMEM * 4);
  float* denom   = (float*)alloc((size_t)BB * NN * 4);

  // ---- one-time prep ----
  {
    int n = BB * NN * NN;
    cvt_bf16_kernel<<<ceil_div(n, 256), 256, 0, stream>>>(adj_in,  adjin_bf,  n);
    cvt_bf16_kernel<<<ceil_div(n, 256), 256, 0, stream>>>(adj_out, adjout_bf, n);
    gcn_denom_kernel<<<4, 256, 0, stream>>>(adj_in, adj_out, denom);
    int np = BB * NN * KP0;
    cvt_pad_kernel<<<ceil_div(np, 256), 256, 0, stream>>>(gcn_inputs, xbf, DIN0, KP0, np);
  }

  for (int l = 0; l < 2; ++l) {
    const int din = (l == 0) ? DIN0 : DMEM;
    const int KP  = (l == 0) ? KP0  : KP1;
    const int NNT = (l == 0) ? NNT0 : NNT1;
    const int NKT = (l == 0) ? NKT0 : NKT1;
    const int bse = 5 + l * 14;
    const float* W[6]  = { (const float*)d_in[bse + 0], (const float*)d_in[bse + 2],
                           (const float*)d_in[bse + 4], (const float*)d_in[bse + 7],
                           (const float*)d_in[bse + 9], (const float*)d_in[bse + 11] };
    const float* bv[6] = { (const float*)d_in[bse + 1], (const float*)d_in[bse + 3],
                           (const float*)d_in[bse + 5], (const float*)d_in[bse + 8],
                           (const float*)d_in[bse + 10], (const float*)d_in[bse + 12] };
    const float* emb   = (const float*)d_in[bse + 6];
    const float* gemb  = (const float*)d_in[bse + 13];

    {
      int n = BB * 8 * NNT * 512;
      pack_x_kernel<<<ceil_div(n, 256), 256, 0, stream>>>(xbf, xpack, KP, NNT);
      int nf = BB * NN * KP;
      fill0_bf16_kernel<<<ceil_div(nf, 256), 256, 0, stream>>>(Axbf, nf);
      fill0_bf16_kernel<<<ceil_div(nf, 256), 256, 0, stream>>>(Bxbf, nf);
      int nw = NKT * NNTW * 512;
      for (int i = 0; i < 6; ++i)
        pack_w_kernel<<<ceil_div(nw, 256), 256, 0, stream>>>(W[i], wp[i], din, NKT);
    }

    if (l == 0) {
      gcn_adj_gemm_kernel<KP0, NNT0><<<dim3(NNT0, 16, BB), 32, 0, stream>>>(
          adjin_bf, adjout_bf, xpack, Axbf, Bxbf);
      gcn_lin3_kernel<KP0, NKT0><<<dim3(NNTW, 64), 32, 0, stream>>>(
          Axbf, Bxbf, xbf, wp[0], wp[1], wp[2], wp[3], wp[4], wp[5],
          bv[0], bv[1], bv[2], bv[3], bv[4], bv[5], denom, resPre, gresPre);
    } else {
      gcn_adj_gemm_kernel<KP1, NNT1><<<dim3(NNT1, 16, BB), 32, 0, stream>>>(
          adjin_bf, adjout_bf, xpack, Axbf, Bxbf);
      gcn_lin3_kernel<KP1, NKT1><<<dim3(NNTW, 64), 32, 0, stream>>>(
          Axbf, Bxbf, xbf, wp[0], wp[1], wp[2], wp[3], wp[4], wp[5],
          bv[0], bv[1], bv[2], bv[3], bv[4], bv[5], denom, resPre, gresPre);
    }

    gcn_bias_emb_kernel<<<dim3(NN, BB), 256, 0, stream>>>(adj_in, adj_out, dep,
                                                          emb, gemb, biasR, biasG);
    int ng = BB * NN * KP1;
    gcn_gate_kernel<<<ceil_div(ng, 256), 256, 0, stream>>>(resPre, gresPre,
                                                           biasR, biasG, xbf);
  }

  {
    int nw = NKT1 * NNTW * 512;
    pack_w_kernel<<<ceil_div(nw, 256), 256, 0, stream>>>(w_mlp, wmlp_p, DMEM, NKT1);
    gcn_mlp_kernel<<<dim3(NNTW, 64), 32, 0, stream>>>(xbf, wmlp_p, b_mlp, (float*)d_out);
  }
}
